// QuantumInspiredLayer_16767552324222
// MI455X (gfx1250) — matrix operand order repaired and checked by reading, not compiled, not run on hardware
//
#include <hip/hip_runtime.h>
#include <hip/hip_bf16.h>
#include <math.h>

// ---------------------------------------------------------------------------
// QuantumInspiredLayer for MI455X (gfx1250, wave32, WMMA).
//   out = sqrt( (X @ cos(theta+tn))^2 + (X @ sin(phi+pn))^2 )
//   X: 16384 x 1024 f32, weights: 1024 x 1024 f32 -> out: 16384 x 1024 f32
// v2: double-buffered LDS staging, SADDR+voffset addressing (one shared
// 32-bit offset register per operand stream, immediates for tile offsets).
// ---------------------------------------------------------------------------

typedef _Float16 v16h __attribute__((ext_vector_type(16)));
typedef _Float16 v8h  __attribute__((ext_vector_type(8)));
typedef _Float16 v4h  __attribute__((ext_vector_type(4)));
typedef float    v8f  __attribute__((ext_vector_type(8)));
typedef float    v4f  __attribute__((ext_vector_type(4)));

#define DIM_K 1024
#define DIM_N 1024
#define DIM_M 16384
#define TILE_M 128
#define TILE_N 128
#define BUF_H  (TILE_M * 32)          // halfs per LDS buffer (8 KB)

// ---------------------------------------------------------------------------
// Kernel 1: build f16 weight matrices in WMMA-B-fragment-friendly layout.
// Packed layout: W[((k/16)*1024 + n)*16 + (k%16)]  -> the 16 f16 values a
// B-fragment lane needs (one n, sixteen consecutive k) are contiguous (32B).
// ---------------------------------------------------------------------------
__global__ __launch_bounds__(256)
void qil_prep_weights(const float* __restrict__ theta,
                      const float* __restrict__ phi,
                      const float* __restrict__ tnoise,
                      const float* __restrict__ pnoise,
                      _Float16* __restrict__ Wc,
                      _Float16* __restrict__ Ws) {
    int idx = blockIdx.x * 256 + threadIdx.x;      // 0 .. 1024*1024-1
    int k = idx >> 10;                             // K index (theta row)
    int n = idx & 1023;                            // N index (theta col)
    float c = cosf(theta[idx] + tnoise[idx]);
    float s = sinf(phi[idx]   + pnoise[idx]);
    int p = (((k >> 4) << 10) + n) * 16 + (k & 15);
    Wc[p] = (_Float16)c;
    Ws[p] = (_Float16)s;
}

// ---------------------------------------------------------------------------
// Kernel 2: fused dual GEMM + magnitude.
// 256 threads = 8 wave32 per workgroup; 128x128 output tile.
// Waves 4(M) x 2(N); wave tile 32x64 = 2x4 WMMA tiles x {real,imag}.
// ---------------------------------------------------------------------------
__global__ __launch_bounds__(256)
void qil_gemm_mag(const float* __restrict__ x,
                  const _Float16* __restrict__ Wc,
                  const _Float16* __restrict__ Ws,
                  float* __restrict__ out) {
    // Double-buffered X slice, staged as f16: 2 x (128 rows x 32 k).
    __shared__ _Float16 xs[2][BUF_H];              // 2 x 8 KB

    const int tid   = threadIdx.x;
    const int lane  = tid & 31;
    const int wid   = tid >> 5;                    // 0..7
    const int wm    = wid & 3;                     // wave M position 0..3
    const int wn    = wid >> 2;                    // wave N position 0..1
    const int lhalf = lane >> 4;                   // 0: lanes 0-15, 1: 16-31
    const int l16   = lane & 15;

    const int m_blk = blockIdx.x * TILE_M;
    const int n_blk = blockIdx.y * TILE_N;

    v8f acc_r[2][4], acc_i[2][4];
    #pragma unroll
    for (int mt = 0; mt < 2; ++mt)
        #pragma unroll
        for (int nt = 0; nt < 4; ++nt) {
            acc_r[mt][nt] = (v8f)0.0f;
            acc_i[mt][nt] = (v8f)0.0f;
        }

    // ---- cooperative staging map: thread t -> one row, 16 f32 (4x float4)
    const int s_row = tid >> 1;                    // 0..127
    const int s_c4  = (tid & 1) * 4;               // float4 chunk base (0 or 4)
    // 32-bit byte offset into x for this thread's first float4 of slice k=0
    unsigned xoff = (unsigned)(m_blk + s_row) * (DIM_K * 4) + (unsigned)s_c4 * 16;
    // per-thread LDS store base (halfs)
    _Float16* sdst = &xs[0][s_row * 32 + s_c4 * 4];

    // ---- per-lane weight byte offset (shared by Wc/Ws and all 4 n-tiles):
    //   byte_off = ((k0/16 + lhalf)*1024 + n_lane) * 32 ; n-tile adds nt*512
    unsigned woff = ((unsigned)lhalf * DIM_N +
                     (unsigned)(n_blk + wn * 64 + l16)) * 32u;

    // ---- per-lane A-fragment LDS base (halfs); mt adds 512 halfs,
    //      buffer adds BUF_H halfs, hi-chunk adds 8 halfs (all immediates)
    const _Float16* abase = &xs[0][(wm * 32 + l16) * 32 + 16 * lhalf];

    // ================= prologue: stage slice k=0 into buffer 0 ============
    {
        #pragma unroll
        for (int j = 0; j < 4; ++j) {
            v4f v = *(const v4f*)((const char*)x + xoff + j * 16);
            v4h h;
            h[0] = (_Float16)v[0]; h[1] = (_Float16)v[1];
            h[2] = (_Float16)v[2]; h[3] = (_Float16)v[3];
            *(v4h*)(sdst + j * 4) = h;
        }
        xoff += 32 * 4;                            // advance one k-slice
    }

    // ================= main loop: 16 iterations x 2 phases ================
    for (int i = 0; i < 16; ++i) {
        #pragma unroll
        for (int p = 0; p < 2; ++p) {
            const int k0 = i * 64 + p * 32;        // slice in buffer p
            const bool stage_next = (k0 + 32) < DIM_K;

            __syncthreads();                       // buffer p staged & safe

            // -- issue next slice's global loads early (hidden under WMMA)
            v4f xv[4];
            if (stage_next) {
                #pragma unroll
                for (int j = 0; j < 4; ++j)
                    xv[j] = *(const v4f*)((const char*)x + xoff + j * 16);
                xoff += 32 * 4;
            }

            // -- B fragments: SADDR-base + shared voffset, imm = nt*512(+16)
            v16h bc[4], bs[4];
            #pragma unroll
            for (int nt = 0; nt < 4; ++nt) {
                bc[nt] = *(const v16h*)((const char*)Wc + woff + nt * 512);
                bs[nt] = *(const v16h*)((const char*)Ws + woff + nt * 512);
            }
            woff += 65536u;                        // next 32-k slice (2 kb rows)

            // -- A fragments from LDS buffer p
            // A 16x32 in this code's contiguous k order (same as B): lanes 0-15 hold K 0-15,
            // lanes 16-31 hold K 16-31 for the same M rows.
            v16h a[2];
            #pragma unroll
            for (int mt = 0; mt < 2; ++mt) {
                const _Float16* ab = abase + p * BUF_H + mt * 512;
                v8h lo = *(const v8h*)(ab);
                v8h hi = *(const v8h*)(ab + 8);
                a[mt] = __builtin_shufflevector(lo, hi,
                         0, 1, 2, 3, 4, 5, 6, 7,
                         8, 9, 10, 11, 12, 13, 14, 15);
            }

            // -- 16 WMMAs: 2 m-tiles x 4 n-tiles x {real, imag}
            #pragma unroll
            for (int mt = 0; mt < 2; ++mt)
                #pragma unroll
                for (int nt = 0; nt < 4; ++nt) {
                    acc_r[mt][nt] = __builtin_amdgcn_wmma_f32_16x16x32_f16(
                        false, a[mt], false, bc[nt], (short)0,
                        acc_r[mt][nt], false, false);
                    acc_i[mt][nt] = __builtin_amdgcn_wmma_f32_16x16x32_f16(
                        false, a[mt], false, bs[nt], (short)0,
                        acc_i[mt][nt], false, false);
                }

            // -- convert + store next slice into the other buffer
            if (stage_next) {
                _Float16* d = sdst + (1 - p) * BUF_H;
                #pragma unroll
                for (int j = 0; j < 4; ++j) {
                    v4h h;
                    h[0] = (_Float16)xv[j][0]; h[1] = (_Float16)xv[j][1];
                    h[2] = (_Float16)xv[j][2]; h[3] = (_Float16)xv[j][3];
                    *(v4h*)(d + j * 4) = h;
                }
            }
        }
    }

    // ================= epilogue: out = sqrt(real^2 + imag^2) ==============
    // C/D layout: lane l holds N = l&15; VGPR r holds M = r + 8*(l>>4).
    #pragma unroll
    for (int mt = 0; mt < 2; ++mt) {
        #pragma unroll
        for (int nt = 0; nt < 4; ++nt) {
            int col  = n_blk + wn * 64 + nt * 16 + l16;
            int row0 = m_blk + wm * 32 + mt * 16 + 8 * lhalf;
            #pragma unroll
            for (int r = 0; r < 8; ++r) {
                float re = acc_r[mt][nt][r];
                float im = acc_i[mt][nt][r];
                out[(size_t)(row0 + r) * DIM_N + col] =
                    sqrtf(re * re + im * im);
            }
        }
    }
}

// ---------------------------------------------------------------------------
// Host-side launch.
// Inputs (setup_inputs order): x, theta, phi, theta_noise, phi_noise.
// Workspace: Wc (2MB f16) | Ws (2MB f16).
// ---------------------------------------------------------------------------
extern "C" void kernel_launch(void* const* d_in, const int* in_sizes, int n_in,
                              void* d_out, int out_size, void* d_ws, size_t ws_size,
                              hipStream_t stream) {
    const float* x      = (const float*)d_in[0];
    const float* theta  = (const float*)d_in[1];
    const float* phi    = (const float*)d_in[2];
    const float* tnoise = (const float*)d_in[3];
    const float* pnoise = (const float*)d_in[4];

    _Float16* Wc = (_Float16*)d_ws;
    _Float16* Ws = Wc + (size_t)DIM_K * DIM_N;

    qil_prep_weights<<<(DIM_K * DIM_N) / 256, 256, 0, stream>>>(
        theta, phi, tnoise, pnoise, Wc, Ws);

    dim3 grid(DIM_M / TILE_M, DIM_N / TILE_N);   // 128 x 8 workgroups
    qil_gemm_mag<<<grid, 256, 0, stream>>>(x, Wc, Ws, (float*)d_out);
}
